// LSTM_18107582120194
// MI455X (gfx1250) — matrix operand order repaired and checked by reading, not compiled, not run on hardware
//
#include <hip/hip_runtime.h>

#define B_SZ   128
#define T_SEQ  1024
#define N_IN   512
#define N_H    1024
#define N_OUT  256
#define K_TOT  1536          // N_IN + N_H, combined GEMM depth per gate
#define KITERS 48            // K_TOT / 32

typedef __bf16 bf16_t;
typedef bf16_t v16bf __attribute__((ext_vector_type(16)));
typedef float  v8f   __attribute__((ext_vector_type(8)));

union Frag {
  uint4 q[2];
  v16bf v;
};

static __device__ __forceinline__ unsigned short f2bf(float f) {
  union { float f; unsigned int u; } x; x.f = f;
  unsigned int u = x.u;
  unsigned int r = u + 0x7FFFu + ((u >> 16) & 1u);   // round-to-nearest-even
  return (unsigned short)(r >> 16);
}

#if __has_builtin(__builtin_amdgcn_cvt_pk_bf16_f32)
static __device__ __forceinline__ unsigned int pk2(float a, float b) {
  auto r = __builtin_amdgcn_cvt_pk_bf16_f32(a, b);
  return __builtin_bit_cast(unsigned int, r);
}
#else
static __device__ __forceinline__ unsigned int pk2(float a, float b) {
  return (unsigned int)f2bf(a) | ((unsigned int)f2bf(b) << 16);
}
#endif

static __device__ __forceinline__ float sigm(float x) {
  return 1.0f / (1.0f + __expf(-x));
}

// scheduling-directive masks (LLVM amdgcn_sched_group_barrier)
#define SGB_WMMA   0x008
#define SGB_DSREAD 0x100

// ---------------- setup kernels ----------------

// in: fp32 [K][N] row-major  ->  out bf16 transposed block: out[n*ldo + koff + k] = in[k][n]
__global__ void k_prep_w(const float* __restrict__ in, unsigned short* __restrict__ out,
                         int K, int N, int ldo, int koff) {
  int idx = blockIdx.x * blockDim.x + threadIdx.x;
  if (idx >= K * N) return;
  int n = idx / K, k = idx - n * K;
  out[(size_t)n * ldo + koff + k] = f2bf(in[(size_t)k * N + n]);
}

__global__ void k_init_h(const float* __restrict__ h0, float* __restrict__ hf,
                         unsigned short* __restrict__ hb) {
  int idx = blockIdx.x * blockDim.x + threadIdx.x;
  if (idx >= B_SZ * N_H) return;
  float v = h0[idx & (N_H - 1)];
  hf[idx] = v;
  hb[idx] = f2bf(v);
}

// ---------------- per-timestep gate kernel ----------------
// grid 16 WGs x 256 threads. WG owns 64 hidden columns, all 4 gates, all 128 batch rows.
// Combined GEMM: [u_t ; h] (K=1536) x Wc[g] for g in {f,i,o,c}.
// Wave w owns batch rows [16w,16w+16); accumulators: 4 gates x 4 n-tiles.
__global__ __launch_bounds__(256)
void k_gates(const float* __restrict__ u,
             const unsigned short* __restrict__ Wc0, const unsigned short* __restrict__ Wc1,
             const unsigned short* __restrict__ Wc2, const unsigned short* __restrict__ Wc3,
             const float* __restrict__ bfv, const float* __restrict__ biv,
             const float* __restrict__ bov, const float* __restrict__ bcv,
             const float* __restrict__ hin_f, const unsigned short* __restrict__ hin_b,
             float* __restrict__ hout_f, unsigned short* __restrict__ hout_b,
             int t)
{
  // ping-pong weight staging: [buf][gate][ntile][n][k], 32 KB
  __shared__ __align__(16) unsigned short sB[2][4][4][16][32];

  const int tid   = threadIdx.x;
  const int wave  = tid >> 5;
  const int lane  = tid & 31;
  const int l16   = lane & 15;
  const int lhalf = lane >> 4;
  const int chunk = blockIdx.x * 64;

  const unsigned short* Wc[4] = {Wc0, Wc1, Wc2, Wc3};

  // LDS staging coords: 256 threads, one 64-byte weight row each
  const int sg  = tid >> 6;
  const int snt = (tid >> 4) & 3;
  const int sn  = tid & 15;
  const unsigned short* srcRow = Wc[sg] + (size_t)(chunk + snt * 16 + sn) * K_TOT;

  auto stage = [&](int it) {
    const unsigned short* s = srcRow + it * 32;
    uint4 a = *(const uint4*)(s);
    uint4 b = *(const uint4*)(s + 8);
    uint4 c = *(const uint4*)(s + 16);
    uint4 d = *(const uint4*)(s + 24);
    *(uint4*)&sB[it & 1][sg][snt][sn][0] = a;
    *(uint4*)&sB[it & 1][sg][snt][sn][8] = b;
    *(uint4*)&sB[it & 1][sg][snt][sn][16] = c;
    *(uint4*)&sB[it & 1][sg][snt][sn][24] = d;
    if (it + 2 < KITERS)
      __builtin_prefetch(srcRow + (it + 2) * 32, 0, 0);   // global_prefetch_b8
  };

  const v8f vzero = {};
  v8f acc[4][4];
#pragma unroll
  for (int g = 0; g < 4; ++g)
#pragma unroll
    for (int nt = 0; nt < 4; ++nt)
      acc[g][nt] = vzero;

  const int arow = 16 * wave + l16;                        // batch row for A-frag
  const float*          uRow = u     + ((size_t)arow * T_SEQ + t) * N_IN;
  const unsigned short* hRow = hin_b + (size_t)arow * N_H;

  stage(0);
  for (int it = 0; it < KITERS; ++it) {
    __syncthreads();                        // sB[it&1] ready; sB[(it+1)&1] free
    if (it + 1 < KITERS) stage(it + 1);

    // A fragment for this K-slice (ISA 16-bit A layout: lane-half K split 0..7 / 16..23)
    Frag af;
    if (it < 16) {                          // u_t contribution (fp32 -> bf16 pack)
      const float* p = uRow + it * 32 + lhalf * 8;
      float4 f0 = *(const float4*)(p);
      float4 f1 = *(const float4*)(p + 4);
      float4 g0 = *(const float4*)(p + 16);
      float4 g1 = *(const float4*)(p + 20);
      af.q[0] = make_uint4(pk2(f0.x, f0.y), pk2(f0.z, f0.w), pk2(f1.x, f1.y), pk2(f1.z, f1.w));
      af.q[1] = make_uint4(pk2(g0.x, g0.y), pk2(g0.z, g0.w), pk2(g1.x, g1.y), pk2(g1.z, g1.w));
    } else {                                // recurrent contribution (bf16 h)
      const unsigned short* p = hRow + (it - 16) * 32 + lhalf * 8;
      af.q[0] = *(const uint4*)(p);
      af.q[1] = *(const uint4*)(p + 16);
    }

    const int buf = it & 1;
#pragma unroll
    for (int g = 0; g < 4; ++g) {
#pragma unroll
      for (int nt = 0; nt < 4; ++nt) {
        Frag bfr;
        const unsigned short* sp = &sB[buf][g][nt][l16][lhalf * 8];
        bfr.q[0] = *(const uint4*)(sp);
        bfr.q[1] = *(const uint4*)(sp + 16);
        acc[g][nt] = __builtin_amdgcn_wmma_f32_16x16x32_bf16(
            false, af.v, false, bfr.v, (short)0, acc[g][nt], false, false);
      }
    }

#if __has_builtin(__builtin_amdgcn_sched_group_barrier)
    // Depth-2 LDS software pipeline:
    //   DSread x4 (frags 0,1) ; 14 x [WMMA x1, DSread x2] ; WMMA x2
    // => every WMMA has the next-but-one fragment's loads already in flight.
    __builtin_amdgcn_sched_group_barrier(SGB_DSREAD, 4, 0);
#pragma unroll
    for (int i = 0; i < 14; ++i) {
      __builtin_amdgcn_sched_group_barrier(SGB_WMMA, 1, 0);
      __builtin_amdgcn_sched_group_barrier(SGB_DSREAD, 2, 0);
    }
    __builtin_amdgcn_sched_group_barrier(SGB_WMMA, 2, 0);
#endif
  }

  // ---- epilogue: bias + gates + state update (c_new = f*h + i*c~ ; c is never read)
#pragma unroll
  for (int nt = 0; nt < 4; ++nt) {
    const int j = chunk + nt * 16 + l16;
    const float Bf = bfv[j], Bi = biv[j], Bo = bov[j], Bc = bcv[j];
#pragma unroll
    for (int v = 0; v < 8; ++v) {
      const int b = 16 * wave + v + 8 * lhalf;
      const size_t o = (size_t)b * N_H + j;
      float fg = sigm(acc[0][nt][v] + Bf);
      float ig = sigm(acc[1][nt][v] + Bi);
      float og = sigm(acc[2][nt][v] + Bo);
      float ct = tanhf(acc[3][nt][v] + Bc);
      float hold = hin_f[o];
      float cn = fg * hold + ig * ct;
      float hn = og * tanhf(cn);
      hout_f[o] = hn;
      hout_b[o] = f2bf(hn);
    }
  }
}

// ---------------- per-timestep output head: y_t = softmax(h_new @ W_hy + b_y) ----------------
// grid 8 WGs x 256 threads. WG owns 16 batch rows; wave w owns columns [32w, 32w+32).
__global__ __launch_bounds__(256)
void k_out(const unsigned short* __restrict__ hnew_b,
           const unsigned short* __restrict__ WtHy,
           const float* __restrict__ by,
           float* __restrict__ out, int t)
{
  __shared__ float sm[16][N_OUT];   // 16 KB

  const int tid   = threadIdx.x;
  const int wave  = tid >> 5;
  const int lane  = tid & 31;
  const int l16   = lane & 15;
  const int lhalf = lane >> 4;
  const int mrow  = blockIdx.x * 16;

  const v8f vzero = {};
  v8f acc[2] = {vzero, vzero};

  const unsigned short* hRow  = hnew_b + (size_t)(mrow + l16) * N_H;
  const unsigned short* wRow0 = WtHy + (size_t)(wave * 32 + l16) * N_H + lhalf * 16;
  const unsigned short* wRow1 = wRow0 + (size_t)16 * N_H;

  for (int kk = 0; kk < N_H; kk += 32) {
    Frag af, b0, b1;
    const unsigned short* p = hRow + kk + lhalf * 16;
    af.q[0] = *(const uint4*)(p);
    af.q[1] = *(const uint4*)(p + 8);
    b0.q[0] = *(const uint4*)(wRow0 + kk);
    b0.q[1] = *(const uint4*)(wRow0 + kk + 8);
    b1.q[0] = *(const uint4*)(wRow1 + kk);
    b1.q[1] = *(const uint4*)(wRow1 + kk + 8);
    acc[0] = __builtin_amdgcn_wmma_f32_16x16x32_bf16(
        false, af.v, false, b0.v, (short)0, acc[0], false, false);
    acc[1] = __builtin_amdgcn_wmma_f32_16x16x32_bf16(
        false, af.v, false, b1.v, (short)0, acc[1], false, false);
  }

#pragma unroll
  for (int nt = 0; nt < 2; ++nt) {
    const int n = wave * 32 + nt * 16 + l16;
    const float Bn = by[n];
#pragma unroll
    for (int v = 0; v < 8; ++v)
      sm[v + 8 * lhalf][n] = acc[nt][v] + Bn;
  }
  __syncthreads();

  if (tid < 16) {
    const int r = tid;
    const int b = mrow + r;
    float mx = -3.402823466e38f;
    for (int n = 0; n < N_OUT; ++n) mx = fmaxf(mx, sm[r][n]);
    float s = 0.0f;
    for (int n = 0; n < N_OUT; ++n) s += __expf(sm[r][n] - mx);
    const float inv = 1.0f / s;
    float* op = out + ((size_t)b * T_SEQ + t) * N_OUT;
    for (int n = 0; n < N_OUT; ++n) op[n] = __expf(sm[r][n] - mx) * inv;
  }
}

// ---------------- host launcher ----------------
extern "C" void kernel_launch(void* const* d_in, const int* in_sizes, int n_in,
                              void* d_out, int out_size, void* d_ws, size_t ws_size,
                              hipStream_t stream) {
  (void)in_sizes; (void)n_in; (void)out_size; (void)ws_size;

  const float* u   = (const float*)d_in[0];
  const float* h0  = (const float*)d_in[15];
  const float* bfv = (const float*)d_in[10];
  const float* biv = (const float*)d_in[11];
  const float* bov = (const float*)d_in[12];
  const float* bcv = (const float*)d_in[13];
  const float* byv = (const float*)d_in[14];
  float* out = (float*)d_out;

  // workspace layout
  char* ws = (char*)d_ws;
  size_t off = 0;
  float* hF[2];
  unsigned short* hB[2];
  hF[0] = (float*)(ws + off); off += (size_t)B_SZ * N_H * sizeof(float);
  hF[1] = (float*)(ws + off); off += (size_t)B_SZ * N_H * sizeof(float);
  hB[0] = (unsigned short*)(ws + off); off += (size_t)B_SZ * N_H * sizeof(unsigned short);
  hB[1] = (unsigned short*)(ws + off); off += (size_t)B_SZ * N_H * sizeof(unsigned short);
  unsigned short* WcG[4];          // combined [N_H][K_TOT] bf16 per gate (3 MB each)
  for (int g = 0; g < 4; ++g) { WcG[g] = (unsigned short*)(ws + off); off += (size_t)N_H * K_TOT * 2; }
  unsigned short* WtHy = (unsigned short*)(ws + off); off += (size_t)N_OUT * N_H * 2;

  // one-time weight transpose/convert into combined [u;h] layout + state init
  for (int g = 0; g < 4; ++g) {
    k_prep_w<<<(N_IN * N_H + 255) / 256, 256, 0, stream>>>(
        (const float*)d_in[1 + g], WcG[g], N_IN, N_H, K_TOT, 0);
    k_prep_w<<<(N_H * N_H + 255) / 256, 256, 0, stream>>>(
        (const float*)d_in[5 + g], WcG[g], N_H, N_H, K_TOT, N_IN);
  }
  k_prep_w<<<(N_H * N_OUT + 255) / 256, 256, 0, stream>>>(
      (const float*)d_in[9], WtHy, N_H, N_OUT, N_H, 0);
  k_init_h<<<(B_SZ * N_H + 255) / 256, 256, 0, stream>>>(h0, hF[0], hB[0]);

  // sequential scan over T; h ping-pongs between parity buffers
  for (int t = 0; t < T_SEQ; ++t) {
    const int pi = t & 1, po = (t + 1) & 1;
    k_gates<<<16, 256, 0, stream>>>(u,
        WcG[0], WcG[1], WcG[2], WcG[3],
        bfv, biv, bov, bcv,
        hF[pi], hB[pi], hF[po], hB[po], t);
    k_out<<<8, 256, 0, stream>>>(hB[po], WtHy, byv, out, t);
  }
}
